// MultiHeadAttention_1073741824647
// MI455X (gfx1250) — matrix operand order repaired and checked by reading, not compiled, not run on hardware
//
#include <hip/hip_runtime.h>
#include <hip/hip_bf16.h>

#define S_LEN 2048
#define HID   4096
#define NH    32
#define NKV   8
#define DH    128
#define ATT_SCALE 0.08838834764831845f   // 1/sqrt(128)

typedef __attribute__((ext_vector_type(16))) _Float16 v16h;
typedef __attribute__((ext_vector_type(8)))  _Float16 v8h;
typedef __attribute__((ext_vector_type(8)))  float    v8f;

// ---------------------------------------------------------------------------
// WMMA helpers (gfx1250: 16x16x32 f16 -> f32)
// ---------------------------------------------------------------------------
__device__ __forceinline__ v8f wmma_f16(v16h a, v16h b, v8f c) {
  return __builtin_amdgcn_wmma_f32_16x16x32_f16(false, a, false, b, (short)0, c,
                                                false, false);
}

// A fragment (16xK tile, K-step 32), A row-major with leading dim lda.
// Per ISA: lanes 0-15 hold K {0..7,16..23}, lanes 16-31 hold K {8..15,24..31};
// both are two contiguous 8-halfword (16B) runs -> 2x B128 loads (global or ds).
__device__ __forceinline__ v16h load_afrag(const _Float16* base, int lda,
                                           int row0, int k0) {
  const int lane = threadIdx.x & 31;
  const int r  = row0 + (lane & 15);
  const int kh = (lane < 16) ? 0 : 8;
  const _Float16* p = base + (size_t)r * lda + k0 + kh;
  v8h lo = *reinterpret_cast<const v8h*>(p);
  v8h hi = *reinterpret_cast<const v8h*>(p + 16);
  v16h f;
#pragma unroll
  for (int i = 0; i < 8; ++i) { f[i] = lo[i]; f[8 + i] = hi[i]; }
  return f;
}

// B fragment (Kx16 tile, K-step 32) where element (k,n) = base[n*ldk + k]
// (i.e. B stored column-major / "transposed weights"). Lanes 0-15: K {0..7,16..23},
// lanes 16-31: K {8..15,24..31} -> two 8-halfword runs -> 2x B128 loads.
__device__ __forceinline__ v16h load_bfrag(const _Float16* base, int ldk,
                                           int col0, int k0) {
  const int lane = threadIdx.x & 31;
  const int c  = col0 + (lane & 15);
  const int kh = (lane < 16) ? 0 : 8;
  const _Float16* p = base + (size_t)c * ldk + k0 + kh;
  v8h lo = *reinterpret_cast<const v8h*>(p);
  v8h hi = *reinterpret_cast<const v8h*>(p + 16);
  v16h f;
#pragma unroll
  for (int i = 0; i < 8; ++i) { f[i] = lo[i]; f[8 + i] = hi[i]; }
  return f;
}

// ---------------------------------------------------------------------------
// f32 -> f16 cast
// ---------------------------------------------------------------------------
__global__ void cvt_kernel(const float* __restrict__ src,
                           _Float16* __restrict__ dst, int n) {
  int i = blockIdx.x * blockDim.x + threadIdx.x;
  if (i < n) dst[i] = (_Float16)src[i];
}

// ---------------------------------------------------------------------------
// LDS-tiled transpose + cast: W f32 [K][N] -> Wt f16 [N][K]
// ---------------------------------------------------------------------------
__global__ void __launch_bounds__(256)
wtrans_kernel(const float* __restrict__ W, _Float16* __restrict__ Wt,
              int K, int N) {
  __shared__ float tile[32][33];
  const int tx = threadIdx.x, ty = threadIdx.y;      // (32, 8)
  const int n0 = blockIdx.x * 32, k0 = blockIdx.y * 32;
#pragma unroll
  for (int j = 0; j < 4; ++j)
    tile[ty + j * 8][tx] = W[(size_t)(k0 + ty + j * 8) * N + (n0 + tx)];
  __syncthreads();
#pragma unroll
  for (int j = 0; j < 4; ++j)
    Wt[(size_t)(n0 + ty + j * 8) * K + (k0 + tx)] =
        (_Float16)tile[tx][ty + j * 8];
}

// ---------------------------------------------------------------------------
// WMMA GEMM: C[M][N] = A[M][K] * Bt[N][K]^T.  128 threads = 4 waves, each
// wave computes a 32x32 tile -> 64x64 per workgroup.
// A tile (64x128 f16, 16KB) is staged to LDS with the gfx1250 async path
// (GLOBAL_LOAD_ASYNC_TO_LDS_B128, ASYNCcnt), then A-fragments come from LDS
// (ds_load_b128); B stays on direct global_load_b128 (weights are L2-resident).
// ---------------------------------------------------------------------------
template <bool F16OUT>
__global__ void __launch_bounds__(128)
gemm_wmma(const _Float16* __restrict__ A, const _Float16* __restrict__ Bt,
          void* __restrict__ C, int M, int N, int K) {
  __shared__ __align__(16) _Float16 Ash[64 * 128];   // 16 KB staging tile
  const int wave = threadIdx.x >> 5;
  const int lane = threadIdx.x & 31;
  const int tn  = blockIdx.x * 64 + (wave & 1) * 32;
  const int tm0 = blockIdx.y * 64;
  const int wm  = (wave >> 1) * 32;                  // wave's rows inside tile
  const unsigned lbase = (unsigned)(size_t)(&Ash[0]);

  v8f acc[2][2] = {};
  for (int kt = 0; kt < K; kt += 128) {
    // ---- async stage A[tm0..tm0+63][kt..kt+127] into LDS ------------------
#pragma unroll
    for (int j = 0; j < 8; ++j) {
      const int sg  = threadIdx.x * 8 + j;           // 0..1023 16B segments
      const int row = sg >> 4;                       // 64 rows
      const int kp  = sg & 15;                       // 16 segments per row
      const _Float16* gp = A + (size_t)(tm0 + row) * K + kt + kp * 8;
      const unsigned lds = lbase + (unsigned)sg * 16u;
      asm volatile("global_load_async_to_lds_b128 %0, %1, off"
                   :: "v"(lds), "v"(gp) : "memory");
    }
    asm volatile("s_wait_asynccnt 0" ::: "memory");
    __syncthreads();
    if (kt + 256 < K)   // gfx1250 global_prefetch path for the weight stream
      __builtin_prefetch(Bt + (size_t)(tn + (lane & 15)) * K + kt + 256, 0, 1);
    // ---- 4 WMMA k-substeps from LDS ---------------------------------------
#pragma unroll
    for (int ks = 0; ks < 4; ++ks) {
      v16h a0 = load_afrag(Ash, 128, wm,      ks * 32);
      v16h a1 = load_afrag(Ash, 128, wm + 16, ks * 32);
      v16h b0 = load_bfrag(Bt, K, tn,      kt + ks * 32);
      v16h b1 = load_bfrag(Bt, K, tn + 16, kt + ks * 32);
      acc[0][0] = wmma_f16(a0, b0, acc[0][0]);
      acc[0][1] = wmma_f16(a0, b1, acc[0][1]);
      acc[1][0] = wmma_f16(a1, b0, acc[1][0]);
      acc[1][1] = wmma_f16(a1, b1, acc[1][1]);
    }
    asm volatile("s_wait_dscnt 0" ::: "memory");     // frags landed in VGPRs
    __syncthreads();                                  // safe to overwrite Ash
  }
  const int roff = (lane < 16) ? 0 : 8;
  const int cc = lane & 15;
#pragma unroll
  for (int mi = 0; mi < 2; ++mi)
#pragma unroll
    for (int ni = 0; ni < 2; ++ni)
#pragma unroll
      for (int v = 0; v < 8; ++v) {
        size_t idx =
            (size_t)(tm0 + wm + mi * 16 + v + roff) * N + (tn + ni * 16 + cc);
        if (F16OUT) ((_Float16*)C)[idx] = (_Float16)acc[mi][ni][v];
        else        ((float*)C)[idx]    = acc[mi][ni][v];
      }
}

// ---------------------------------------------------------------------------
// RoPE + relayout: lin f16 [S][nh*128] -> out f16 [nh][S][128]
// cos/sin are f32 [64][S] (reference layout (B, D/2, S))
// ---------------------------------------------------------------------------
__global__ void rope_kernel(const _Float16* __restrict__ lin,
                            const float* __restrict__ cp,
                            const float* __restrict__ sp,
                            _Float16* __restrict__ out, int nh) {
  const int idx = blockIdx.x * blockDim.x + threadIdx.x;   // nh*S*64 threads
  const int i = idx & 63;
  const int s = (idx >> 6) & (S_LEN - 1);
  const int h = idx >> 17;
  if (h >= nh) return;
  const _Float16* pin = lin + (size_t)s * (nh * DH) + h * DH + 2 * i;
  const float x1 = (float)pin[0];
  const float x2 = (float)pin[1];
  const float c  = cp[i * S_LEN + s];
  const float sn = sp[i * S_LEN + s];
  _Float16* po = out + ((size_t)h * S_LEN + s) * DH + 2 * i;
  po[0] = (_Float16)(x1 * c - x2 * sn);
  po[1] = (_Float16)(x1 * sn + x2 * c);
}

// ---------------------------------------------------------------------------
// V transpose: Vlin f16 [S][NKV*128] -> Vt f16 [NKV][128][S]
// ---------------------------------------------------------------------------
__global__ void vtrans_kernel(const _Float16* __restrict__ Vlin,
                              _Float16* __restrict__ Vt) {
  const int idx = blockIdx.x * blockDim.x + threadIdx.x;   // NKV*128*S threads
  const int s  = idx & (S_LEN - 1);
  const int d  = (idx >> 11) & 127;
  const int hv = idx >> 18;
  Vt[idx] = Vlin[(size_t)s * (NKV * DH) + hv * DH + d];
}

// ---------------------------------------------------------------------------
// Flash attention: 4 waves/block, each wave owns one 16-query tile of one
// head. Online softmax over 32-key chunks; causal mask applied analytically
// (== reference's -1e9 additive mask). P goes through a per-wave LDS tile to
// convert C-layout scores into A-layout fragments.
// ---------------------------------------------------------------------------
__global__ void __launch_bounds__(128)
attn_fa(const _Float16* __restrict__ Qr, const _Float16* __restrict__ Kr,
        const _Float16* __restrict__ Vt, _Float16* __restrict__ Ao) {
  __shared__ __align__(16) _Float16 Plds[4][16][40];   // 40 = padded stride
  const int wave = threadIdx.x >> 5;
  const int lane = threadIdx.x & 31;
  const int h   = blockIdx.y;
  const int hkv = h >> 2;                               // N_REP = 4
  const int qb  = blockIdx.x * 64 + wave * 16;
  const _Float16* Qb = Qr + ((size_t)h * S_LEN + qb) * DH;
  const _Float16* Kb = Kr + (size_t)hkv * S_LEN * DH;
  const _Float16* Vb = Vt + (size_t)hkv * DH * S_LEN;

  v16h qf[4];
#pragma unroll
  for (int dc = 0; dc < 4; ++dc) qf[dc] = load_afrag(Qb, DH, 0, dc * 32);

  v8f o[8] = {};
  float m[8], l[8];
#pragma unroll
  for (int v = 0; v < 8; ++v) { m[v] = -1e30f; l[v] = 0.f; }

  const int roff = (lane < 16) ? 0 : 8;
  const int cc = lane & 15;
  const int nch = (qb + 47) >> 5;                       // ceil((qb+16)/32)

  for (int ch = 0; ch < nch; ++ch) {
    const int kb = ch << 5;
    v8f s0 = {}, s1 = {};
#pragma unroll
    for (int dc = 0; dc < 4; ++dc) {
      v16h b0 = load_bfrag(Kb, DH, kb,      dc * 32);
      v16h b1 = load_bfrag(Kb, DH, kb + 16, dc * 32);
      s0 = wmma_f16(qf[dc], b0, s0);
      s1 = wmma_f16(qf[dc], b1, s1);
    }
    float p0[8], p1[8], al[8];
#pragma unroll
    for (int v = 0; v < 8; ++v) {
      const int row = qb + v + roff;
      float a = s0[v] * ATT_SCALE + ((kb + cc)      > row ? -1e9f : 0.f);
      float b = s1[v] * ATT_SCALE + ((kb + 16 + cc) > row ? -1e9f : 0.f);
      float cm = fmaxf(a, b);
#pragma unroll
      for (int msk = 1; msk < 16; msk <<= 1)
        cm = fmaxf(cm, __shfl_xor(cm, msk, 32));
      const float mn = fmaxf(m[v], cm);
      al[v] = __expf(m[v] - mn);
      p0[v] = __expf(a - mn);
      p1[v] = __expf(b - mn);
      float rs = p0[v] + p1[v];
#pragma unroll
      for (int msk = 1; msk < 16; msk <<= 1)
        rs += __shfl_xor(rs, msk, 32);
      l[v] = l[v] * al[v] + rs;
      m[v] = mn;
    }
#pragma unroll
    for (int dc = 0; dc < 8; ++dc)
#pragma unroll
      for (int v = 0; v < 8; ++v) o[dc][v] *= al[v];
    // C-layout P -> LDS -> A-layout fragment (wave-private, DS pipe in-order)
#pragma unroll
    for (int v = 0; v < 8; ++v) {
      const int rp = v + roff;
      Plds[wave][rp][cc]      = (_Float16)p0[v];
      Plds[wave][rp][cc + 16] = (_Float16)p1[v];
    }
    asm volatile("s_wait_dscnt 0" ::: "memory");
    v16h pf = load_afrag(&Plds[wave][0][0], 40, 0, 0);
#pragma unroll
    for (int dc = 0; dc < 8; ++dc) {
      v16h vb = load_bfrag(Vb, S_LEN, dc * 16, kb);
      o[dc] = wmma_f16(pf, vb, o[dc]);
    }
    asm volatile("" ::: "memory");   // keep LDS reuse ordered across chunks
  }
#pragma unroll
  for (int v = 0; v < 8; ++v) l[v] = 1.f / l[v];
#pragma unroll
  for (int dc = 0; dc < 8; ++dc)
#pragma unroll
    for (int v = 0; v < 8; ++v) {
      const int q = qb + v + roff;
      Ao[(size_t)q * HID + h * DH + dc * 16 + cc] =
          (_Float16)(o[dc][v] * l[v]);
    }
}

// ---------------------------------------------------------------------------
extern "C" void kernel_launch(void* const* d_in, const int* in_sizes, int n_in,
                              void* d_out, int out_size, void* d_ws,
                              size_t ws_size, hipStream_t stream) {
  (void)in_sizes; (void)n_in; (void)out_size; (void)ws_size;
  const float* hs = (const float*)d_in[0];
  const float* cp = (const float*)d_in[1];
  const float* sp = (const float*)d_in[2];
  // d_in[3] (attention_mask) is reproduced analytically (-1e9 causal add)
  const float* Wq = (const float*)d_in[4];
  const float* Wk = (const float*)d_in[5];
  const float* Wv = (const float*)d_in[6];
  const float* Wo = (const float*)d_in[7];

  _Float16* ws = (_Float16*)d_ws;
  size_t off = 0;
  _Float16* Xh   = ws + off; off += (size_t)S_LEN * HID;       // 16 MB
  _Float16* WqT  = ws + off; off += (size_t)HID * HID;         // 32 MB
  _Float16* WkT  = ws + off; off += (size_t)HID * (NKV * DH);  //  8 MB
  _Float16* WvT  = ws + off; off += (size_t)HID * (NKV * DH);  //  8 MB
  _Float16* WoT  = ws + off; off += (size_t)HID * HID;         // 32 MB
  _Float16* Qlin = ws + off; off += (size_t)S_LEN * HID;       // 16 MB
  _Float16* Klin = ws + off; off += (size_t)S_LEN * (NKV * DH);
  _Float16* Vlin = ws + off; off += (size_t)S_LEN * (NKV * DH);
  _Float16* Qr   = ws + off; off += (size_t)NH * S_LEN * DH;   // 16 MB
  _Float16* Kr   = ws + off; off += (size_t)NKV * S_LEN * DH;
  _Float16* Vt   = ws + off; off += (size_t)NKV * DH * S_LEN;
  _Float16* Ao   = Qlin;   // Qlin dead after RoPE; reuse for attention output

  // 1) cast activations to f16
  cvt_kernel<<<(S_LEN * HID) / 256, 256, 0, stream>>>(hs, Xh, S_LEN * HID);

  // 2) transpose+cast all weights to f16 [N][K]
  dim3 tb(32, 8);
  wtrans_kernel<<<dim3(HID / 32, HID / 32), tb, 0, stream>>>(Wq, WqT, HID, HID);
  wtrans_kernel<<<dim3((NKV * DH) / 32, HID / 32), tb, 0, stream>>>(Wk, WkT, HID, NKV * DH);
  wtrans_kernel<<<dim3((NKV * DH) / 32, HID / 32), tb, 0, stream>>>(Wv, WvT, HID, NKV * DH);
  wtrans_kernel<<<dim3(HID / 32, HID / 32), tb, 0, stream>>>(Wo, WoT, HID, HID);

  // 3) QKV projections (WMMA, f16 out)
  gemm_wmma<true><<<dim3(HID / 64, S_LEN / 64), 128, 0, stream>>>(
      Xh, WqT, Qlin, S_LEN, HID, HID);
  gemm_wmma<true><<<dim3((NKV * DH) / 64, S_LEN / 64), 128, 0, stream>>>(
      Xh, WkT, Klin, S_LEN, NKV * DH, HID);
  gemm_wmma<true><<<dim3((NKV * DH) / 64, S_LEN / 64), 128, 0, stream>>>(
      Xh, WvT, Vlin, S_LEN, NKV * DH, HID);

  // 4) RoPE + head-major relayout; V transpose to [hkv][d][s]
  rope_kernel<<<(NH * S_LEN * 64) / 256, 256, 0, stream>>>(Qlin, cp, sp, Qr, NH);
  rope_kernel<<<(NKV * S_LEN * 64) / 256, 256, 0, stream>>>(Klin, cp, sp, Kr, NKV);
  vtrans_kernel<<<(NKV * DH * S_LEN) / 256, 256, 0, stream>>>(Vlin, Vt);

  // 5) flash attention (WMMA for QK^T and PV)
  attn_fa<<<dim3(S_LEN / 64, NH), 128, 0, stream>>>(Qr, Kr, Vt, Ao);

  // 6) output projection -> f32 d_out
  gemm_wmma<false><<<dim3(HID / 64, S_LEN / 64), 128, 0, stream>>>(
      Ao, WoT, d_out, S_LEN, HID, HID);
}
